// GGSUBase_21199958573751
// MI455X (gfx1250) — compile-verified, matrix operand order checked by reading, not run on hardware
//
#include <hip/hip_runtime.h>
#include <hip/hip_bf16.h>

typedef __attribute__((ext_vector_type(16))) _Float16 v16h;
typedef __attribute__((ext_vector_type(8)))  float    v8f;

#define Bn   8
#define CIN  128
#define Hh   128
#define Ww   128
#define HW   16384
#define Gg   8

__constant__ int d_SH[4] = {1, 11, 3, 5};
__constant__ int d_SW[4] = {11, 1, 5, 3};

// A/B fragment K index for v_wmma_f32_16x16x32_f16 (wave32):
// lane<16: elems 0..7 -> K=0..7, elems 8..15 -> K=16..23
// lane>=16: elems 0..7 -> K=8..15, elems 8..15 -> K=24..31
__device__ __forceinline__ int kmap(int e, int lane) {
    return e + (e & 8) + ((lane & 16) >> 1);
}

__device__ __forceinline__ float fast_rcp(float x) {
    return __builtin_amdgcn_rcpf(x);      // single v_rcp_f32
}

// Bilinear-weight A-fragment, specialized per branch so div/mod/compares
// constant-fold (br is wave-uniform -> scalar branch dispatch).
template <int SHB, int SWB>
__device__ __forceinline__ v16h make_Au(int lane, int px, int iy0, float wy) {
    constexpr int NC = SHB * SWB;
    int ix0 = 0; float wx = 0.f;
    if (SWB > 1) {
        float srcx = (float)px * (float)(SWB - 1) * (1.0f / 127.0f);
        ix0 = (int)floorf(srcx);
        if (ix0 > SWB - 2) ix0 = SWB - 2;
        wx = srcx - (float)ix0;
    }
    v16h Au;
#pragma unroll
    for (int e = 0; e < 16; ++e) {
        int k = kmap(e, lane);
        float v = 0.f;
        if (k < NC) {
            int cy = k / SWB, cx = k % SWB;          // compile-time strength-reduced
            float uy = (SHB == 1) ? 1.f
                     : (cy == iy0 ? 1.f - wy : (cy == iy0 + 1 ? wy : 0.f));
            float ux = (SWB == 1) ? 1.f
                     : (cx == ix0 ? 1.f - wx : (cx == ix0 + 1 ? wx : 0.f));
            v = uy * ux;
        }
        Au[e] = (_Float16)v;
    }
    return Au;
}

// ---------------------------------------------------------------------------
// Kernel 1: adaptive-avg-pool (exact torch bins) + per-branch linear layer.
// One block per (b, c). Output: tbl[b][br][c][16] (cells padded with zeros,
// linear bias already folded in — bilinear weights sum to 1 so bias is exact).
// ---------------------------------------------------------------------------
__global__ void pool_table_kernel(const float* __restrict__ x,
                                  const float* __restrict__ W0, const float* __restrict__ b0,
                                  const float* __restrict__ W1, const float* __restrict__ b1,
                                  const float* __restrict__ W2, const float* __restrict__ b2,
                                  const float* __restrict__ W3, const float* __restrict__ b3,
                                  float* __restrict__ tbl)
{
    int bc = blockIdx.x;
    int b = bc >> 7, c = bc & 127;
    const float* img = x + (size_t)(b * CIN + c) * HW;

    __shared__ float rowpool[4][128][12];   // [branch][row][w-bin]
    __shared__ float pooled[4][16];

    int t = threadIdx.x;                    // 128 threads, t = row
    {
        int h = t;
        for (int br = 0; br < 4; ++br) {
            int swb = d_SW[br];
            for (int cx = 0; cx < swb; ++cx) {
                int s = (cx * Ww) / swb;
                int e = ((cx + 1) * Ww + swb - 1) / swb;
                float acc = 0.f;
                for (int w = s; w < e; ++w) acc += img[h * Ww + w];
                rowpool[br][h][cx] = acc / (float)(e - s);
            }
        }
    }
    __syncthreads();

    if (t < 64) {
        int br = t >> 4, cell = t & 15;
        int shb = d_SH[br], swb = d_SW[br];
        float v = 0.f;
        if (cell < shb * swb) {
            int cy = cell / swb, cx = cell % swb;
            int s = (cy * Hh) / shb;
            int e = ((cy + 1) * Hh + shb - 1) / shb;
            float acc = 0.f;
            for (int h = s; h < e; ++h) acc += rowpool[br][h][cx];
            v = acc / (float)(e - s);
        }
        pooled[br][cell] = v;
    }
    __syncthreads();

    if (t < 64) {
        int br = t >> 4, cell = t & 15;
        const float* Wl = (br == 0) ? W0 : (br == 1) ? W1 : (br == 2) ? W2 : W3;
        const float* bl = (br == 0) ? b0 : (br == 1) ? b1 : (br == 2) ? b2 : b3;
        int cd = d_SH[br] * d_SW[br];
        float v = 0.f;
        if (cell < cd) {
            v = bl[cell];
            for (int j = 0; j < cd; ++j) v += Wl[cell * cd + j] * pooled[br][j];
        }
        tbl[((size_t)(b * 4 + br) * 128 + c) * 16 + cell] = v;
    }
}

// ---------------------------------------------------------------------------
// Kernel 2: Wc2[gr][o][i] = sum_h Whh[gr,o,h] * W_center[gr,h,i]   (96x16/group)
// Folds hidden-path GEMM so it runs off the raw x fragment (K=16).
// ---------------------------------------------------------------------------
__global__ void prep_wc2_kernel(const float* __restrict__ Wc,
                                const float* __restrict__ Whh,
                                float* __restrict__ wc2)
{
    int idx = blockIdx.x * 256 + threadIdx.x;   // 8*96*16 = 12288
    if (idx >= Gg * 96 * 16) return;
    int i  = idx & 15;
    int o  = (idx >> 4) % 96;
    int gr = idx / (96 * 16);
    float acc = 0.f;
    for (int h = 0; h < 32; ++h)
        acc += Whh[(gr * 96 + o) * 32 + h] * Wc[(gr * 32 + h) * 16 + i];
    wc2[idx] = acc;
}

// ---------------------------------------------------------------------------
// Kernel 3: Cmat[b][gr][o][cell] = sum_f Wih[gr,o,f] * Bmat[cell,f]
//   Bmat[cell,f] = sum_i Wm[g,f&1,i] * tbl[b][gr>>1][c0+i][cell],
//   g = gr*16 + (f>>1), c0 = (4g) & 127.
// Folds border->merge->gi input GEMM into one 96x16 matrix per (b,group).
// ---------------------------------------------------------------------------
__global__ void prep_cmat_kernel(const float* __restrict__ Wih,
                                 const float* __restrict__ Wm,
                                 const float* __restrict__ tbl,
                                 float* __restrict__ cmat)
{
    int idx = blockIdx.x * 256 + threadIdx.x;   // 8*8*96*16 = 98304
    if (idx >= Bn * Gg * 96 * 16) return;
    int cell = idx & 15;
    int o    = (idx >> 4) % 96;
    int gr   = (idx / (16 * 96)) & 7;
    int b    = idx / (16 * 96 * 8);
    int br   = gr >> 1;
    const float* tb = tbl + (size_t)(b * 4 + br) * 128 * 16;
    float acc = 0.f;
    for (int f = 0; f < 32; ++f) {
        int g  = gr * 16 + (f >> 1);
        int o2 = f & 1;
        int c0 = (4 * g) & 127;
        float bm = 0.f;
        for (int i = 0; i < 4; ++i)
            bm += Wm[g * 8 + o2 * 4 + i] * tb[(c0 + i) * 16 + cell];
        acc += Wih[(gr * 96 + o) * 32 + f] * bm;
    }
    cmat[idx] = acc;
}

// ---------------------------------------------------------------------------
// Kernel 4: fused main kernel. Block = (b,row), 8 waves = 8 GRU groups,
// 8 tiles of 16 pixels. 14 v_wmma_f32_16x16x32_f16 per tile per wave.
// Reads x once (coalesced), writes out once; no HBM intermediates.
// ---------------------------------------------------------------------------
__global__ void __launch_bounds__(256)
main_fused_kernel(const float* __restrict__ x,
                  const float* __restrict__ Wc,
                  const float* __restrict__ wc2,
                  const float* __restrict__ cmat,
                  const float* __restrict__ bih,
                  const float* __restrict__ bhh,
                  float* __restrict__ out)
{
    int blk  = blockIdx.x;
    int b    = blk >> 7;
    int row  = blk & 127;
    int gr   = threadIdx.x >> 5;     // wave = GRU group
    int lane = threadIdx.x & 31;
    int br   = gr >> 1;
    int shb  = d_SH[br];
    int lcol = lane & 15;

    // ---- weight B-fragments (per-wave constants, hoisted out of tile loop)
    v16h Bc[2], Bc2[6], Bcm[6];
#pragma unroll
    for (int t = 0; t < 2; ++t)
#pragma unroll
        for (int e = 0; e < 16; ++e) {
            int k = kmap(e, lane);
            int col = t * 16 + lcol;
            Bc[t][e] = (k < 16) ? (_Float16)Wc[(gr * 32 + col) * 16 + k] : (_Float16)0.f;
        }
#pragma unroll
    for (int t = 0; t < 6; ++t)
#pragma unroll
        for (int e = 0; e < 16; ++e) {
            int k = kmap(e, lane);
            int col = t * 16 + lcol;
            Bc2[t][e] = (k < 16) ? (_Float16)wc2[(gr * 96 + col) * 16 + k] : (_Float16)0.f;
            Bcm[t][e] = (k < 16) ? (_Float16)cmat[(((size_t)b * 8 + gr) * 96 + col) * 16 + k]
                                 : (_Float16)0.f;
        }

    // ---- y-axis bilinear factors (constant per block row, per branch)
    int iy0 = 0; float wy = 0.f;
    if (shb > 1) {
        float srcy = (float)row * (float)(shb - 1) * (1.0f / 127.0f);
        iy0 = (int)floorf(srcy);
        if (iy0 > shb - 2) iy0 = shb - 2;
        wy = srcy - (float)iy0;
    }

    // ---- gate biases (column = 16*tq + lcol), hoisted
    float b_r[2], b_z[2], b_in[2], b_hn[2];
#pragma unroll
    for (int tq = 0; tq < 2; ++tq) {
        int col = tq * 16 + lcol;
        b_r[tq]  = bih[gr * 96 + col]      + bhh[gr * 96 + col];
        b_z[tq]  = bih[gr * 96 + 32 + col] + bhh[gr * 96 + 32 + col];
        b_in[tq] = bih[gr * 96 + 64 + col];
        b_hn[tq] = bhh[gr * 96 + 64 + col];
    }

    const v8f vzero = {0.f, 0.f, 0.f, 0.f, 0.f, 0.f, 0.f, 0.f};
    size_t xbase = ((size_t)b * CIN + gr * 16) * HW + (size_t)row * Ww;

    for (int t8 = 0; t8 < 8; ++t8) {
        int wbase = t8 * 16;
        int px = wbase + lcol;              // this lane's pixel (M = lcol)

        // A-fragment of x: 16 pixels x 16 channels (K 16..31 zero-padded)
        v16h Ax;
#pragma unroll
        for (int e = 0; e < 8; ++e) {
            int k = kmap(e, lane);          // in [0,16)
            Ax[e] = (_Float16)x[xbase + (size_t)k * HW + px];
        }
#pragma unroll
        for (int e = 8; e < 16; ++e) Ax[e] = (_Float16)0.f;

        // A-fragment of bilinear weights U (branch-specialized, br wave-uniform)
        v16h Au;
        switch (br) {
            case 0:  Au = make_Au<1, 11>(lane, px, iy0, wy); break;
            case 1:  Au = make_Au<11, 1>(lane, px, iy0, wy); break;
            case 2:  Au = make_Au<3, 5>(lane, px, iy0, wy); break;
            default: Au = make_Au<5, 3>(lane, px, iy0, wy); break;
        }

        // ---- WMMAs ----
        v8f cg[2], arz[4], ahn[2], ain[2];
#pragma unroll
        for (int t = 0; t < 2; ++t)   // hidden state (needed for blend)
            cg[t] = __builtin_amdgcn_wmma_f32_16x16x32_f16(false, Ax, false, Bc[t],
                                                           (short)0, vzero, false, false);
#pragma unroll
        for (int t = 0; t < 4; ++t)   // h_r, h_z  (folded x @ Wc2)
            arz[t] = __builtin_amdgcn_wmma_f32_16x16x32_f16(false, Ax, false, Bc2[t],
                                                            (short)0, vzero, false, false);
#pragma unroll
        for (int q = 0; q < 2; ++q)   // h_n
            ahn[q] = __builtin_amdgcn_wmma_f32_16x16x32_f16(false, Ax, false, Bc2[4 + q],
                                                            (short)0, vzero, false, false);
#pragma unroll
        for (int t = 0; t < 4; ++t)   // i_r, i_z accumulate onto h_r, h_z
            arz[t] = __builtin_amdgcn_wmma_f32_16x16x32_f16(false, Au, false, Bcm[t],
                                                            (short)0, arz[t], false, false);
#pragma unroll
        for (int q = 0; q < 2; ++q)   // i_n
            ain[q] = __builtin_amdgcn_wmma_f32_16x16x32_f16(false, Au, false, Bcm[4 + q],
                                                            (short)0, vzero, false, false);

        // ---- gates + blend + store (C layout: elem = pixel, lane = column)
        // sigmoid/tanh via v_exp_f32 + v_rcp_f32 (no IEEE div sequences)
#pragma unroll
        for (int tq = 0; tq < 2; ++tq) {
            int col = tq * 16 + lcol;   // hidden index h
            float yb[8];
#pragma unroll
            for (int e = 0; e < 8; ++e) {
                float rr = fast_rcp(1.f + __expf(-(arz[tq][e]     + b_r[tq])));
                float zz = fast_rcp(1.f + __expf(-(arz[2 + tq][e] + b_z[tq])));
                float a2 = ain[tq][e] + b_in[tq] + rr * (ahn[tq][e] + b_hn[tq]);
                float nn = 1.f - 2.f * fast_rcp(__expf(2.f * a2) + 1.f);   // tanh
                yb[e] = (1.f - zz) * nn + zz * cg[tq][e];
            }
            // out channel = h*8 + gr ; pixels = wbase + (lane<16 ? 0..7 : 8..15)
            size_t base = ((size_t)b * 256 + col * 8 + gr) * HW
                        + (size_t)row * Ww + wbase + ((lane >> 4) << 3);
            float4* o4 = reinterpret_cast<float4*>(out + base);
            o4[0] = make_float4(yb[0], yb[1], yb[2], yb[3]);
            o4[1] = make_float4(yb[4], yb[5], yb[6], yb[7]);
        }
    }
}

// ---------------------------------------------------------------------------
extern "C" void kernel_launch(void* const* d_in, const int* in_sizes, int n_in,
                              void* d_out, int out_size, void* d_ws, size_t ws_size,
                              hipStream_t stream) {
    const float* x   = (const float*)d_in[0];
    const float* Wc  = (const float*)d_in[1];
    const float* W0  = (const float*)d_in[2];
    const float* b0  = (const float*)d_in[3];
    const float* W1  = (const float*)d_in[4];
    const float* b1  = (const float*)d_in[5];
    const float* W2  = (const float*)d_in[6];
    const float* b2  = (const float*)d_in[7];
    const float* W3  = (const float*)d_in[8];
    const float* b3  = (const float*)d_in[9];
    const float* Wm  = (const float*)d_in[10];
    const float* Wih = (const float*)d_in[11];
    const float* Whh = (const float*)d_in[12];
    const float* bih = (const float*)d_in[13];
    const float* bhh = (const float*)d_in[14];
    float* out = (float*)d_out;

    // workspace layout (f32): tables | Wc2 | Cmat
    float* tbl  = (float*)d_ws;                 // 8*4*128*16 = 65536
    float* wc2  = tbl + 65536;                  // 8*96*16    = 12288
    float* cmat = wc2 + 12288;                  // 8*8*96*16  = 98304

    pool_table_kernel<<<dim3(Bn * CIN), dim3(128), 0, stream>>>(
        x, W0, b0, W1, b1, W2, b2, W3, b3, tbl);
    prep_wc2_kernel<<<dim3(48), dim3(256), 0, stream>>>(Wc, Whh, wc2);
    prep_cmat_kernel<<<dim3(384), dim3(256), 0, stream>>>(Wih, Wm, tbl, cmat);
    main_fused_kernel<<<dim3(Bn * Hh), dim3(256), 0, stream>>>(
        x, Wc, wc2, cmat, bih, bhh, out);
}
